// HQVAE_13262859010640
// MI455X (gfx1250) — matrix operand order repaired and checked by reading, not compiled, not run on hardware
//
#include <hip/hip_runtime.h>
#include <hip/hip_bf16.h>

// ---------------------------------------------------------------------------
// HQ-VAE forward for MI455X (gfx1250): bf16 WMMA GEMM pipeline, f32 accumulate.
// Data movement: double-buffered LDS tiles filled with gfx1250 async copies
// (global_load_async_to_lds_b128, ASYNCcnt), no VGPR round-trip.
// All weights are pre-transposed to [N,K] bf16 so every tile is a contiguous-K
// 16B-chunk copy (A activations are already [M,K] row-major).
// ---------------------------------------------------------------------------

typedef __bf16 bf16;
typedef __attribute__((ext_vector_type(16))) __bf16 v16bf;
typedef __attribute__((ext_vector_type(8)))  float  v8f;

#define BM 128
#define BN 128
#define BK 32
#define SPITCH 48   // bf16 elements; 96B row pitch -> 16B aligned fragment reads

union FragAB { v16bf v; float4 f4[2]; };

__device__ __forceinline__ unsigned ford_u32(float f) {
  unsigned u = __float_as_uint(f);
  return (u & 0x80000000u) ? ~u : (u | 0x80000000u);  // order-preserving map
}

__device__ __forceinline__ unsigned long long shfl_xor_u64(unsigned long long v, int mask) {
  unsigned lo = (unsigned)v, hi = (unsigned)(v >> 32);
  lo = __shfl_xor(lo, mask, 32);
  hi = __shfl_xor(hi, mask, 32);
  return ((unsigned long long)hi << 32) | lo;
}

// LDS aperture puts the allocation-relative byte offset in addr[31:0] (ISA 10.2)
__device__ __forceinline__ unsigned lds_off(const void* p) {
  return (unsigned)(unsigned long long)(size_t)p;
}

__device__ __forceinline__ void async_ld_b128(unsigned ldst, const void* gsrc) {
  asm volatile("global_load_async_to_lds_b128 %0, %1, off"
               :: "v"(ldst), "v"(gsrc) : "memory");
}

// ---------------------------------------------------------------------------
// GEMM: C[M,N] = act(A[M,K] * B^T + bias); A bf16 [M,K] (lda), B bf16 [N,K] (ldb)
// ARGMIN: no C store; per-row argmin of (cbnorm[n] - 2*acc) via atomicMin(u64)
// ---------------------------------------------------------------------------
template <bool RELU, bool STORE_BF, bool STORE_F32, bool ARGMIN>
__global__ __launch_bounds__(256) void hqvae_gemm_wmma(
    const bf16* __restrict__ A, int lda,
    const bf16* __restrict__ Bm, int ldb,
    const float* __restrict__ bias,
    bf16* __restrict__ Cbf, int ldcb,
    float* __restrict__ Cf, int ldcf,
    const float* __restrict__ cbnorm,
    unsigned long long* __restrict__ rowmin,
    int N, int K)
{
  __shared__ bf16 sA [2][BM][SPITCH];
  __shared__ bf16 sBT[2][BN][SPITCH];

  const int tid  = threadIdx.x;
  const int lane = tid & 31;
  const int wave = tid >> 5;
  const int lm   = lane & 15;
  const int hi   = lane >> 4;
  const int wm   = wave & 3;     // 4 waves along M
  const int wn   = wave >> 2;    // 2 waves along N
  const int m0   = blockIdx.y * BM;
  const int n0   = blockIdx.x * BN;

  v8f acc[2][4];
#pragma unroll
  for (int i = 0; i < 2; ++i)
#pragma unroll
    for (int j = 0; j < 4; ++j) {
      v8f z = {0.f, 0.f, 0.f, 0.f, 0.f, 0.f, 0.f, 0.f};
      acc[i][j] = z;
    }

  // per call: 4 async b128 copies per thread (2 A-chunks, 2 B-chunks)
  auto stage = [&](int buf, int kt) {
#pragma unroll
    for (int i = 0; i < 2; ++i) {
      int c  = tid + i * 256;          // 512 chunks of 8 bf16 per tile
      int r  = c >> 2;
      int cc = c & 3;
      const bf16* ga = A + (size_t)(m0 + r) * lda + kt + cc * 8;
      async_ld_b128(lds_off(&sA[buf][r][cc * 8]), ga);
      int rn = n0 + r; if (rn >= N) rn = N - 1;   // clamp: cols >= N never stored
      const bf16* gb = Bm + (size_t)rn * ldb + kt + cc * 8;
      async_ld_b128(lds_off(&sBT[buf][r][cc * 8]), gb);
    }
  };

  const int nk = K / BK;
  stage(0, 0);

  for (int t = 0; t < nk; ++t) {
    const int buf = t & 1;
    if (t + 1 < nk) {
      stage(buf ^ 1, (t + 1) * BK);                       // overlap next tile
      asm volatile("s_wait_asynccnt 0x4" ::: "memory");   // tile t landed (in-order)
    } else {
      asm volatile("s_wait_asynccnt 0x0" ::: "memory");
    }
    __syncthreads();   // all waves' copies visible

    FragAB aF[2], bF[4];
#pragma unroll
    for (int tm = 0; tm < 2; ++tm) {
      int r = wm * 32 + tm * 16 + lm;
      aF[tm].f4[0] = *reinterpret_cast<const float4*>(&sA[buf][r][hi * 8]);       // K 0-7 / 8-15
      aF[tm].f4[1] = *reinterpret_cast<const float4*>(&sA[buf][r][16 + hi * 8]);  // K 16-23 / 24-31
    }
#pragma unroll
    for (int tn = 0; tn < 4; ++tn) {
      int c = wn * 64 + tn * 16 + lm;
      bF[tn].f4[0] = *reinterpret_cast<const float4*>(&sBT[buf][c][hi * 8]);      // K 0-7 / 8-15
      bF[tn].f4[1] = *reinterpret_cast<const float4*>(&sBT[buf][c][16 + hi * 8]); // K 16-23 / 24-31
    }
#pragma unroll
    for (int tm = 0; tm < 2; ++tm)
#pragma unroll
      for (int tn = 0; tn < 4; ++tn)
        acc[tm][tn] = __builtin_amdgcn_wmma_f32_16x16x32_bf16(
            false, aF[tm].v, false, bF[tn].v, (short)0, acc[tm][tn], false, false);

    __syncthreads();   // done reading buf before it is re-staged next iteration
  }

  const int gmBase = m0 + wm * 32 + hi * 8;

  if constexpr (ARGMIN) {
#pragma unroll
    for (int tm = 0; tm < 2; ++tm) {
#pragma unroll
      for (int r = 0; r < 8; ++r) {
        unsigned long long best = ~0ULL;
#pragma unroll
        for (int tn = 0; tn < 4; ++tn) {
          int gn = n0 + wn * 64 + tn * 16 + lm;
          unsigned long long key = ~0ULL;
          if (gn < N) {
            float dist = cbnorm[gn] - 2.0f * acc[tm][tn][r];
            key = ((unsigned long long)ford_u32(dist) << 32) | (unsigned)gn;
          }
          best = key < best ? key : best;
        }
#pragma unroll
        for (int msk = 1; msk < 16; msk <<= 1) {
          unsigned long long o = shfl_xor_u64(best, msk);
          best = o < best ? o : best;
        }
        if (lm == 0) atomicMin(rowmin + gmBase + tm * 16 + r, best);
      }
    }
  } else {
#pragma unroll
    for (int tm = 0; tm < 2; ++tm) {
#pragma unroll
      for (int tn = 0; tn < 4; ++tn) {
        int gn = n0 + wn * 64 + tn * 16 + lm;
        if (gn < N) {
          float bv = bias[gn];
#pragma unroll
          for (int r = 0; r < 8; ++r) {
            float v = acc[tm][tn][r] + bv;
            if constexpr (RELU) v = v > 0.f ? v : 0.f;
            size_t gm = (size_t)(gmBase + tm * 16 + r);
            if constexpr (STORE_BF)  Cbf[gm * ldcb + gn] = (bf16)v;
            if constexpr (STORE_F32) Cf [gm * ldcf + gn] = v;
          }
        }
      }
    }
  }
}

// ---------------------------------------------------------------------------
// Small utility kernels
// ---------------------------------------------------------------------------
__global__ __launch_bounds__(256) void hqvae_init(unsigned long long* rowmin, long nrm,
                                                  float* scal, int ns) {
  long i = (long)blockIdx.x * 256 + threadIdx.x;
  if (i < nrm) rowmin[i] = ~0ULL;
  if (i < ns)  scal[i] = 0.f;
}

__global__ __launch_bounds__(256) void hqvae_cvt_bf16(const float* __restrict__ src,
                                                      bf16* __restrict__ dst, long n) {
  long i = (long)blockIdx.x * 256 + threadIdx.x;
  long stride = (long)gridDim.x * 256;
  for (; i < n; i += stride) dst[i] = (bf16)src[i];
}

// src [R,C] f32 row-major -> dst [C,R] bf16 row-major (weight transpose)
__global__ __launch_bounds__(256) void hqvae_cvt_T(const float* __restrict__ src, int R, int C,
                                                   bf16* __restrict__ dst) {
  long n = (long)R * C;
  long i = (long)blockIdx.x * 256 + threadIdx.x;
  long stride = (long)gridDim.x * 256;
  for (; i < n; i += stride) {
    int r = (int)(i / C), c = (int)(i % C);
    dst[(size_t)c * R + r] = (bf16)src[i];
  }
}

// src [R,C] -> dst rows (rowoff..rowoff+C-1) of a [*,ldk] transposed concat
__global__ __launch_bounds__(256) void hqvae_cvt_cols_T(const float* __restrict__ src, int R, int C,
                                                        bf16* __restrict__ dst, int ldk, int rowoff) {
  long n = (long)R * C;
  long i = (long)blockIdx.x * 256 + threadIdx.x;
  long stride = (long)gridDim.x * 256;
  for (; i < n; i += stride) {
    int r = (int)(i / C), c = (int)(i % C);
    dst[(size_t)(rowoff + c) * ldk + r] = (bf16)src[i];
  }
}

__global__ __launch_bounds__(256) void hqvae_copy_f32(const float* __restrict__ src,
                                                      float* __restrict__ dst, int n) {
  int i = blockIdx.x * 256 + threadIdx.x;
  if (i < n) dst[i] = src[i];
}

__global__ __launch_bounds__(256) void hqvae_cbnorm(const float* __restrict__ cb, int Kcb, int d,
                                                    float* __restrict__ out) {
  int n = blockIdx.x * 256 + threadIdx.x;
  if (n < Kcb) {
    float s = 0.f;
    for (int j = 0; j < d; ++j) { float v = cb[(size_t)n * d + j]; s += v * v; }
    out[n] = s;
  }
}

// one wave per row: gather codebook row, write bf16 quantized, accum commit loss
__global__ __launch_bounds__(256) void hqvae_vq_gather(
    const unsigned long long* __restrict__ rowmin,
    const float* __restrict__ cb, int d,
    const float* __restrict__ P, int ldp, int coloff,
    bf16* __restrict__ Q, int ldq,
    float* __restrict__ vq_acc)
{
  int wave = threadIdx.x >> 5, lane = threadIdx.x & 31;
  int m = blockIdx.x * 8 + wave;
  unsigned idx = (unsigned)(rowmin[m] & 0xFFFFFFFFu);
  const float* crow = cb + (size_t)idx * d;
  float s = 0.f;
  for (int j = lane; j < d; j += 32) {
    float q = crow[j];
    float p = P[(size_t)m * ldp + coloff + j];
    float dd = q - p;
    s += dd * dd;
    Q[(size_t)m * ldq + coloff + j] = (bf16)q;
  }
  for (int msk = 16; msk; msk >>= 1) s += __shfl_xor(s, msk, 32);
  if (lane == 0) atomicAdd(vq_acc, s);
}

__global__ __launch_bounds__(256) void hqvae_kl(const float* __restrict__ MULV,
                                                const float* __restrict__ pm,
                                                const float* __restrict__ plv,
                                                float* __restrict__ acc) {
  int wave = threadIdx.x >> 5, lane = threadIdx.x & 31;
  int m = blockIdx.x * 8 + wave;
  float s = 0.f;
  for (int j = lane; j < 128; j += 32) {
    float mu = MULV[(size_t)m * 256 + j];
    float lv = MULV[(size_t)m * 256 + 128 + j];
    float p0 = pm[j], p1 = plv[j];
    float dmu = mu - p0;
    s += p1 - lv + (__expf(lv) + dmu * dmu) * __expf(-p1) - 1.0f;
  }
  for (int msk = 16; msk; msk >>= 1) s += __shfl_xor(s, msk, 32);
  if (lane == 0) atomicAdd(acc, 0.5f * s);
}

__global__ __launch_bounds__(256) void hqvae_sqdiff(const float* __restrict__ a,
                                                    const float* __restrict__ b,
                                                    long n, float* __restrict__ acc) {
  long i = (long)blockIdx.x * 256 + threadIdx.x;
  long stride = (long)gridDim.x * 256;
  float s = 0.f;
  for (; i < n; i += stride) { float d = a[i] - b[i]; s += d * d; }
  for (int msk = 16; msk; msk >>= 1) s += __shfl_xor(s, msk, 32);
  if ((threadIdx.x & 31) == 0) atomicAdd(acc, s);
}

__global__ void hqvae_finalize(const float* __restrict__ scal, float* __restrict__ out_total) {
  float recon = scal[0] / (32768.0f * 1024.0f);
  float kl    = scal[1] / 32768.0f;
  float vq    = 1.25f * (scal[2] / (32768.0f * 64.0f) +
                         scal[3] / (32768.0f * 128.0f) +
                         scal[4] / (32768.0f * 256.0f));
  *out_total = recon + vq + 0.1f * kl;
}

// ---------------------------------------------------------------------------
// Host-side helpers
// ---------------------------------------------------------------------------
static void run_gemm(hipStream_t s, int M, int N, int K,
                     const bf16* A, int lda, const bf16* B, int ldb,
                     const float* bias, bf16* Cbf, int ldcb, float* Cf, int ldcf,
                     bool relu, const float* cbnorm = nullptr,
                     unsigned long long* rowmin = nullptr) {
  dim3 grid((N + BN - 1) / BN, M / BM), blk(256);
  if (rowmin) {
    hqvae_gemm_wmma<false, false, false, true><<<grid, blk, 0, s>>>(
        A, lda, B, ldb, nullptr, nullptr, 0, nullptr, 0, cbnorm, rowmin, N, K);
  } else if (Cbf && Cf) {
    hqvae_gemm_wmma<false, true, true, false><<<grid, blk, 0, s>>>(
        A, lda, B, ldb, bias, Cbf, ldcb, Cf, ldcf, nullptr, nullptr, N, K);
  } else if (Cbf) {
    if (relu)
      hqvae_gemm_wmma<true, true, false, false><<<grid, blk, 0, s>>>(
          A, lda, B, ldb, bias, Cbf, ldcb, nullptr, 0, nullptr, nullptr, N, K);
    else
      hqvae_gemm_wmma<false, true, false, false><<<grid, blk, 0, s>>>(
          A, lda, B, ldb, bias, Cbf, ldcb, nullptr, 0, nullptr, nullptr, N, K);
  } else {
    hqvae_gemm_wmma<false, false, true, false><<<grid, blk, 0, s>>>(
        A, lda, B, ldb, bias, nullptr, 0, Cf, ldcf, nullptr, nullptr, N, K);
  }
}

static void cvt(hipStream_t s, const float* src, bf16* dst, long n) {
  long g = (n + 255) / 256; if (g > 4096) g = 4096;
  hqvae_cvt_bf16<<<(int)g, 256, 0, s>>>(src, dst, n);
}

static void cvtT(hipStream_t s, const float* src, int R, int C, bf16* dst) {
  long n = (long)R * C;
  long g = (n + 255) / 256; if (g > 2048) g = 2048;
  hqvae_cvt_T<<<(int)g, 256, 0, s>>>(src, R, C, dst);
}

extern "C" void kernel_launch(void* const* d_in, const int* in_sizes, int n_in,
                              void* d_out, int out_size, void* d_ws, size_t ws_size,
                              hipStream_t stream) {
  if (n_in < 39) return;
  const int Bn = 32768;
  const int dE[3] = {64, 128, 256}, dH[3] = {32, 64, 128};
  const int Kcb[3] = {512, 1024, 2048};
  const int offE[3] = {0, 64, 192}, offH[3] = {0, 32, 96};

  const float* x     = (const float*)d_in[0];
  const float* enc0W = (const float*)d_in[2];  const float* enc0b = (const float*)d_in[3];
  const float* enc1W = (const float*)d_in[4];  const float* enc1b = (const float*)d_in[5];
  const float* hW[3] = {(const float*)d_in[6], (const float*)d_in[8], (const float*)d_in[10]};
  const float* hb[3] = {(const float*)d_in[7], (const float*)d_in[9], (const float*)d_in[11]};
  const float* muW = (const float*)d_in[12];  const float* mub = (const float*)d_in[13];
  const float* lvW = (const float*)d_in[14];  const float* lvb = (const float*)d_in[15];
  const float* pm  = (const float*)d_in[16];  const float* plv = (const float*)d_in[17];
  const float* p1W[3] = {(const float*)d_in[18], (const float*)d_in[22], (const float*)d_in[26]};
  const float* p1b[3] = {(const float*)d_in[19], (const float*)d_in[23], (const float*)d_in[27]};
  const float* p2W[3] = {(const float*)d_in[20], (const float*)d_in[24], (const float*)d_in[28]};
  const float* p2b[3] = {(const float*)d_in[21], (const float*)d_in[25], (const float*)d_in[29]};
  const float* cb[3]  = {(const float*)d_in[30], (const float*)d_in[31], (const float*)d_in[32]};
  const float* dW[3]  = {(const float*)d_in[33], (const float*)d_in[35], (const float*)d_in[37]};
  const float* db[3]  = {(const float*)d_in[34], (const float*)d_in[36], (const float*)d_in[38]};

  // ---- workspace carve-out ----
  char* ws = (char*)d_ws;
  size_t cur = 0;
  auto carve = [&](size_t bytes) -> char* {
    char* p = ws + cur; cur = (cur + bytes + 255) & ~(size_t)255; return p;
  };
  bf16*  X_BF  = (bf16*)carve((size_t)Bn * 1024 * 2);
  bf16*  H1    = (bf16*)carve((size_t)Bn * 2048 * 2);   // reused as dec hidden 2
  bf16*  H2    = (bf16*)carve((size_t)Bn * 1024 * 2);   // reused as dec hidden 1
  bf16*  FEATS = (bf16*)carve((size_t)Bn * 448 * 2);
  bf16*  PH    = (bf16*)carve((size_t)Bn * 224 * 2);
  bf16*  PBF   = (bf16*)carve((size_t)Bn * 448 * 2);
  float* PF32  = (float*)carve((size_t)Bn * 448 * 4);
  bf16*  QBF   = (bf16*)carve((size_t)Bn * 448 * 2);
  float* MULV  = (float*)carve((size_t)Bn * 256 * 4);
  unsigned long long* ROWMIN = (unsigned long long*)carve((size_t)3 * Bn * 8);
  bf16*  WBE0T = (bf16*)carve((size_t)2048 * 1024 * 2);   // [2048,1024]
  bf16*  WBE1T = (bf16*)carve((size_t)1024 * 2048 * 2);   // [1024,2048]
  bf16*  WBHT  = (bf16*)carve((size_t)448 * 1024 * 2);    // [448,1024]
  float* BH    = (float*)carve(448 * 4);
  bf16*  WBVBT = (bf16*)carve((size_t)256 * 448 * 2);     // [256,448]
  float* BVB   = (float*)carve(256 * 4);
  bf16*  WBP1T[3]; bf16* WBP2T[3]; bf16* CBB[3]; float* CBN[3];
  for (int l = 0; l < 3; ++l) {
    WBP1T[l] = (bf16*)carve((size_t)dH[l] * dE[l] * 2);   // [d/2, d]
    WBP2T[l] = (bf16*)carve((size_t)dE[l] * dH[l] * 2);   // [d, d/2]
    CBB[l]   = (bf16*)carve((size_t)Kcb[l] * dE[l] * 2);  // [Kcb, d] (already N-major)
    CBN[l]   = (float*)carve((size_t)Kcb[l] * 4);
  }
  bf16*  WBD0T = (bf16*)carve((size_t)1024 * 448 * 2);    // [1024,448]
  bf16*  WBD1T = (bf16*)carve((size_t)2048 * 1024 * 2);   // [2048,1024]
  bf16*  WBD2T = (bf16*)carve((size_t)1024 * 2048 * 2);   // [1024,2048]
  float* SCAL  = (float*)carve(8 * 4);   // 0 recon, 1 kl, 2..4 vq per level
  if (cur > ws_size) return;             // workspace too small: bail cleanly

  float* recon = (float*)d_out;
  float* total = recon + (size_t)Bn * 1024;

  // ---- init accumulators / argmin keys ----
  hqvae_init<<<(3 * Bn + 255) / 256, 256, 0, stream>>>(ROWMIN, (long)3 * Bn, SCAL, 8);

  // ---- bf16 conversions; weights transposed to [N,K] ----
  cvt(stream, x, X_BF, (long)Bn * 1024);
  cvtT(stream, enc0W, 1024, 2048, WBE0T);
  cvtT(stream, enc1W, 2048, 1024, WBE1T);
  for (int l = 0; l < 3; ++l) {
    hqvae_cvt_cols_T<<<512, 256, 0, stream>>>(hW[l], 1024, dE[l], WBHT, 1024, offE[l]);
    hqvae_copy_f32<<<2, 256, 0, stream>>>(hb[l], BH + offE[l], dE[l]);
    cvtT(stream, p1W[l], dE[l], dH[l], WBP1T[l]);
    cvtT(stream, p2W[l], dH[l], dE[l], WBP2T[l]);
    cvt(stream, cb[l], CBB[l], (long)Kcb[l] * dE[l]);
    hqvae_cbnorm<<<(Kcb[l] + 255) / 256, 256, 0, stream>>>(cb[l], Kcb[l], dE[l], CBN[l]);
  }
  hqvae_cvt_cols_T<<<256, 256, 0, stream>>>(muW, 448, 128, WBVBT, 448, 0);
  hqvae_cvt_cols_T<<<256, 256, 0, stream>>>(lvW, 448, 128, WBVBT, 448, 128);
  hqvae_copy_f32<<<1, 256, 0, stream>>>(mub, BVB, 128);
  hqvae_copy_f32<<<1, 256, 0, stream>>>(lvb, BVB + 128, 128);
  cvtT(stream, dW[0], 448, 1024, WBD0T);
  cvtT(stream, dW[1], 1024, 2048, WBD1T);
  cvtT(stream, dW[2], 2048, 1024, WBD2T);

  // ---- encoder ----
  run_gemm(stream, Bn, 2048, 1024, X_BF, 1024, WBE0T, 1024, enc0b, H1, 2048, nullptr, 0, true);
  run_gemm(stream, Bn, 1024, 2048, H1, 2048, WBE1T, 2048, enc1b, H2, 1024, nullptr, 0, true);
  // fused heads -> feats [B,448]
  run_gemm(stream, Bn, 448, 1024, H2, 1024, WBHT, 1024, BH, FEATS, 448, nullptr, 0, false);
  // fused mu|logvar -> [B,256] f32
  run_gemm(stream, Bn, 256, 448, FEATS, 448, WBVBT, 448, BVB, nullptr, 0, MULV, 256, false);

  // ---- per-level projection MLP + VQ distance/argmin ----
  for (int l = 0; l < 3; ++l) {
    run_gemm(stream, Bn, dH[l], dE[l], FEATS + offE[l], 448, WBP1T[l], dE[l], p1b[l],
             PH + offH[l], 224, nullptr, 0, true);
    run_gemm(stream, Bn, dE[l], dH[l], PH + offH[l], 224, WBP2T[l], dH[l], p2b[l],
             PBF + offE[l], 448, PF32 + offE[l], 448, false);
    run_gemm(stream, Bn, Kcb[l], dE[l], PBF + offE[l], 448, CBB[l], dE[l], nullptr,
             nullptr, 0, nullptr, 0, false, CBN[l], ROWMIN + (size_t)l * Bn);
  }
  // gather quantized rows + commit loss
  for (int l = 0; l < 3; ++l) {
    hqvae_vq_gather<<<Bn / 8, 256, 0, stream>>>(ROWMIN + (size_t)l * Bn, cb[l], dE[l],
                                                PF32, 448, offE[l], QBF, 448, SCAL + 2 + l);
  }

  // ---- decoder ----
  run_gemm(stream, Bn, 1024, 448, QBF, 448, WBD0T, 448, db[0], H2, 1024, nullptr, 0, true);
  run_gemm(stream, Bn, 2048, 1024, H2, 1024, WBD1T, 1024, db[1], H1, 2048, nullptr, 0, true);
  run_gemm(stream, Bn, 1024, 2048, H1, 2048, WBD2T, 2048, db[2], nullptr, 0, recon, 1024, false);

  // ---- losses ----
  hqvae_sqdiff<<<4096, 256, 0, stream>>>(recon, x, (long)Bn * 1024, SCAL + 0);
  hqvae_kl<<<Bn / 8, 256, 0, stream>>>(MULV, pm, plv, SCAL + 1);
  hqvae_finalize<<<1, 1, 0, stream>>>(SCAL, total);
}
